// CausalSelfAttention_69715909149476
// MI455X (gfx1250) — matrix operand order repaired and checked by reading, not compiled, not run on hardware
//
#include <hip/hip_runtime.h>

// ---------------------------------------------------------------------------
// CausalSelfAttention for MI455X (gfx1250, wave32, WMMA + TDM).
// B=4, T=2048, DIM=1024, H=16, D=64.
// GEMMs + attention via v_wmma_f32_16x16x32_f16; tile staging via
// tensor_load_to_lds (TDM, 6-arg builtin) with TENSORcnt ping-pong.
// ---------------------------------------------------------------------------

typedef _Float16 half_t;
typedef __attribute__((ext_vector_type(16))) _Float16     v16h;
typedef __attribute__((ext_vector_type(8)))  float        v8f;
typedef __attribute__((ext_vector_type(8)))  unsigned int v8u;
typedef __attribute__((ext_vector_type(4)))  unsigned int u32x4;
typedef __attribute__((ext_vector_type(8)))  int          i32x8;
typedef __attribute__((ext_vector_type(4)))  int          i32x4;

#define BATCH 4
#define SEQ   2048
#define EMB   1024
#define NH    16
#define HD    64

#if defined(__has_builtin)
#  if __has_builtin(__builtin_amdgcn_tensor_load_to_lds)
#    define HAVE_TDM 1
#  endif
#endif
#ifndef HAVE_TDM
#  define HAVE_TDM 0
#endif

#if HAVE_TDM
// Build a 2D TDM descriptor (ISA ch.8) and issue TENSOR_LOAD_TO_LDS.
// tileW halves per row, tileH rows, global row stride strideW halves,
// LDS rows padded by 8 halves (pad_amount code 3 = 4 DWORDs);
// padCode: 3 -> pad every 16 DWORDs (32-half rows), 4 -> every 32 (64-half).
__device__ __forceinline__ void tdm_load_2d(const void* gsrc, unsigned ldsoff,
                                            unsigned tileW, unsigned tileH,
                                            unsigned strideW, unsigned padCode) {
    const unsigned long long ga = (unsigned long long)(size_t)gsrc;
    u32x4 g0;
    g0[0] = 1u;                                                  // count=1
    g0[1] = ldsoff;                                              // lds_addr
    g0[2] = (unsigned)ga;                                        // gaddr lo
    g0[3] = ((unsigned)(ga >> 32) & 0x01FFFFFFu) | (2u << 30);   // gaddr hi|type=2
    i32x8 g1;
    g1[0] = (int)((1u << 16) | (1u << 20) | (padCode << 22) | (3u << 25));
    g1[1] = (int)(tileW << 16);        // tensor_dim0[15:0]
    g1[2] = (int)(tileH << 16);        // tensor_dim1[15:0]
    g1[3] = (int)(tileW << 16);        // tile_dim0
    g1[4] = (int)tileH;                // tile_dim1 (tile_dim2 = 0)
    g1[5] = (int)strideW;              // tensor_dim0_stride lo
    g1[6] = 0;
    g1[7] = 0;
    const i32x4 gz4 = {0, 0, 0, 0};
    const i32x8 gz8 = {0, 0, 0, 0, 0, 0, 0, 0};
    __builtin_amdgcn_tensor_load_to_lds(g0, g1, gz4, gz4, gz8, 0);
}
#  if __has_builtin(__builtin_amdgcn_s_wait_tensorcnt)
#    define WAIT_TENSOR() __builtin_amdgcn_s_wait_tensorcnt(0)
#  else
#    define WAIT_TENSOR() asm volatile("s_wait_tensorcnt 0" ::: "memory")
#  endif
#endif

// ---------------- WMMA fragment helpers (CDNA5 wave32 layouts) --------------
__device__ __forceinline__ v16h frag_a(const half_t* base, int ldh) {
    const int lane = threadIdx.x & 31;
    const int m  = lane & 15;
    const int hs = (lane >> 4) & 1;
    const unsigned int* row = (const unsigned int*)(base + (size_t)m * ldh);
    v8u w;
#pragma unroll
    for (int v = 0; v < 4; ++v) {
        w[v]     = row[4 * hs + v];       // K = 8*hs + 2v, 2v+1
        w[v + 4] = row[8 + 4 * hs + v];   // K = 16 + 8*hs + 2v, 2v+1
    }
    return __builtin_bit_cast(v16h, w);
}

__device__ __forceinline__ v16h frag_b(const half_t* base, int ldh) {
    const int lane = threadIdx.x & 31;
    const int n  = lane & 15;
    const int hs = (lane >> 4) & 1;
    const unsigned int* row = (const unsigned int*)(base + (size_t)n * ldh);
    v8u w;
#pragma unroll
    for (int v = 0; v < 4; ++v) {
        w[v]     = row[4 * hs + v];       // K = 8*hs + 2v, 2v+1
        w[v + 4] = row[8 + 4 * hs + v];   // K = 16 + 8*hs + 2v, 2v+1
    }
    return __builtin_bit_cast(v16h, w);
}

__device__ __forceinline__ v8f wmma16(v16h a, v16h b, v8f c) {
    return __builtin_amdgcn_wmma_f32_16x16x32_f16(false, a, false, b,
                                                  (short)0, c, false, false);
}

// ---------------------------- f32 -> f16 convert ----------------------------
__global__ void cvt_f32_f16(const float* __restrict__ in,
                            half_t* __restrict__ out, int n) {
    for (int i = blockIdx.x * blockDim.x + threadIdx.x; i < n;
         i += gridDim.x * blockDim.x)
        out[i] = (half_t)in[i];
}

// --------------------- WMMA GEMM: C[M,N] = A[M,K] * Bw[N,K]^T ---------------
// Block tile 128x256, 256 threads = 8 waves (2x4), wave tile 64x64 (4x4 WMMA).
// Double-buffered LDS; tiles staged by TDM (waves 0/1) when available.
template <bool F16OUT>
__global__ __launch_bounds__(256)
void wmma_gemm_nt(const half_t* __restrict__ A, const half_t* __restrict__ Bw,
                  float* __restrict__ Cf, half_t* __restrict__ Ch,
                  int M, int N, int K) {
    __shared__ half_t Alds[2][128 * 40];
    __shared__ half_t Blds[2][256 * 40];

    const int tid  = threadIdx.x;
    const int lane = tid & 31;
    const int wave = tid >> 5;
    const int bm = (blockIdx.x % (M / 128)) * 128;
    const int bn = (blockIdx.x / (M / 128)) * 256;
    const int wm = (wave >> 2) * 64;   // 0,64
    const int wn = (wave & 3) * 64;    // 0..192

    v8f acc[4][4] = {};

    auto stage = [&](int buf, int kc) {
#if HAVE_TDM
        if (wave == 0)
            tdm_load_2d(A + (size_t)bm * K + kc,
                        (unsigned)(size_t)&Alds[buf][0], 32, 128, (unsigned)K, 3);
        else if (wave == 1)
            tdm_load_2d(Bw + (size_t)bn * K + kc,
                        (unsigned)(size_t)&Blds[buf][0], 32, 256, (unsigned)K, 3);
#else
        {   // A: 128 rows x 32 halves, 2 x uint4 per thread
            const int row = tid >> 1, c = (tid & 1) * 16;
            const uint4* gp = (const uint4*)(A + (size_t)(bm + row) * K + kc + c);
            uint4 v0 = gp[0], v1 = gp[1];
            uint4* lp = (uint4*)(&Alds[buf][row * 40 + c]);
            lp[0] = v0; lp[1] = v1;
        }
        {   // B: 256 rows x 32 halves, 4 x uint4 per thread
            const int row = tid;
            const uint4* gp = (const uint4*)(Bw + (size_t)(bn + row) * K + kc);
            uint4 v0 = gp[0], v1 = gp[1], v2 = gp[2], v3 = gp[3];
            uint4* lp = (uint4*)(&Blds[buf][row * 40]);
            lp[0] = v0; lp[1] = v1; lp[2] = v2; lp[3] = v3;
        }
#endif
    };

    stage(0, 0);
    int buf = 0;
    for (int kc = 0; kc < K; kc += 32) {
#if HAVE_TDM
        if (wave < 2) WAIT_TENSOR();
#endif
        __syncthreads();
        if (kc + 32 < K) stage(buf ^ 1, kc + 32);   // overlap DMA with compute

        // Preload all 8 fragments, then issue 16 independent WMMAs back-to-back
        v16h af[4], bf[4];
#pragma unroll
        for (int i = 0; i < 4; ++i)
            af[i] = frag_a(&Alds[buf][(wm + i * 16) * 40], 40);
#pragma unroll
        for (int j = 0; j < 4; ++j)
            bf[j] = frag_b(&Blds[buf][(wn + j * 16) * 40], 40);
#pragma unroll
        for (int j = 0; j < 4; ++j)
#pragma unroll
            for (int i = 0; i < 4; ++i)
                acc[i][j] = wmma16(af[i], bf[j], acc[i][j]);
        buf ^= 1;
    }

    const int n  = lane & 15;
    const int hs = lane >> 4;
#pragma unroll
    for (int i = 0; i < 4; ++i)
#pragma unroll
        for (int j = 0; j < 4; ++j)
#pragma unroll
            for (int r = 0; r < 8; ++r) {
                const int row = bm + wm + i * 16 + r + 8 * hs;
                const int col = bn + wn + j * 16 + n;
                if (F16OUT) Ch[(size_t)row * N + col] = (half_t)acc[i][j][r];
                else        Cf[(size_t)row * N + col] = acc[i][j][r];
            }
}

// ---------------- RoPE + reshape qkv[B,T,3*EMB] -> q/k/v [B,H,T,D] ----------
__global__ void rope_split(const half_t* __restrict__ qkv,
                           half_t* __restrict__ qh, half_t* __restrict__ kh,
                           half_t* __restrict__ vh) {
    int idx = blockIdx.x * blockDim.x + threadIdx.x;     // 1<<22 total
    if (idx >= (BATCH * NH * SEQ * (HD / 2))) return;
    const int d = idx & 31;
    const int t = (idx >> 5) & (SEQ - 1);
    const int h = (idx >> 16) & (NH - 1);
    const int b = idx >> 20;

    const size_t ib = ((size_t)(b * SEQ + t)) * (3 * EMB) + h * HD + d;
    const float q1 = (float)qkv[ib],          q2 = (float)qkv[ib + 32];
    const float k1 = (float)qkv[ib + EMB],    k2 = (float)qkv[ib + EMB + 32];
    const half_t v1 = qkv[ib + 2 * EMB],      v2 = qkv[ib + 2 * EMB + 32];

    const float inv = __powf(10000.f, -(float)d * (1.f / 32.f));
    float s, c;
    __sincosf((float)t * inv, &s, &c);

    const size_t ob = (((size_t)(b * NH + h)) * SEQ + t) * HD + d;
    qh[ob]      = (half_t)(q1 * c - q2 * s);
    qh[ob + 32] = (half_t)(q2 * c + q1 * s);
    kh[ob]      = (half_t)(k1 * c - k2 * s);
    kh[ob + 32] = (half_t)(k2 * c + k1 * s);
    vh[ob]      = v1;
    vh[ob + 32] = v2;
}

// --------------------------- Flash attention --------------------------------
// 128 threads = 4 waves, each wave owns 16 query rows (block = 64 queries).
// K tile staged by TDM; V transposed into LDS; online softmax with the
// row-sum computed by an extra WMMA against a ones matrix.
// Causal structure: waves skip fully-masked blocks entirely; masking/guards
// only on the single diagonal block per wave.
__global__ __launch_bounds__(128)
void flash_attn(const half_t* __restrict__ qh, const half_t* __restrict__ kh,
                const half_t* __restrict__ vh, half_t* __restrict__ yh) {
    __shared__ half_t Klds[32 * 72];     // [key][d], +8-half pad
    __shared__ half_t VldsT[64 * 40];    // [d][key], +8-half pad
    __shared__ half_t Plds[4][16 * 40];  // per-wave P tile [m][key]

    const int tid  = threadIdx.x;
    const int lane = tid & 31;
    const int wave = tid >> 5;
    const int nQB  = SEQ / 64;
    const int qb   = (blockIdx.x % nQB) * 64;
    const int bh   = blockIdx.x / nQB;           // b*NH + h
    const int b    = bh >> 4, h = bh & 15;
    const size_t base = (size_t)bh * SEQ * HD;
    const int qbase = qb + wave * 16;
    const int n  = lane & 15;
    const int hs = lane >> 4;

    // Q fragments, pre-scaled by 1/sqrt(D) (packed f16 muls, done once)
    const half_t* Qtile = qh + base + (size_t)qbase * HD;
    v16h qf0 = frag_a(Qtile, HD);
    v16h qf1 = frag_a(Qtile + 32, HD);
#pragma unroll
    for (int v = 0; v < 16; ++v) {
        qf0[v] = qf0[v] * (half_t)0.125f;
        qf1[v] = qf1[v] * (half_t)0.125f;
    }

    v16h onesf;
#pragma unroll
    for (int v = 0; v < 16; ++v) onesf[v] = (half_t)1.0f;

    float mrow[8], lrow[8];
    v8f acc[4] = {};
#pragma unroll
    for (int r = 0; r < 8; ++r) { mrow[r] = -3.0e38f; lrow[r] = 0.f; }

    const int qmaxwg = qb + 63;
    for (int kb = 0; kb <= qmaxwg; kb += 32) {
#if HAVE_TDM
        if (wave == 0)
            tdm_load_2d(kh + base + (size_t)kb * HD,
                        (unsigned)(size_t)&Klds[0], 64, 32, HD, 4);
#endif
        // V transposed into LDS (element scatter; manual — TDM can't transpose)
#pragma unroll
        for (int i = 0; i < 2; ++i) {
            const int idx = tid + 128 * i;
            const int row = idx >> 3, c = idx & 7;
#if !HAVE_TDM
            uint4 kv = *(const uint4*)(kh + base + (size_t)(kb + row) * HD + c * 8);
            *(uint4*)(Klds + row * 72 + c * 8) = kv;
#endif
            uint4 vv = *(const uint4*)(vh + base + (size_t)(kb + row) * HD + c * 8);
            const half_t* hp = (const half_t*)&vv;
#pragma unroll
            for (int j = 0; j < 8; ++j) VldsT[(c * 8 + j) * 40 + row] = hp[j];
        }
#if HAVE_TDM
        if (wave == 0) WAIT_TENSOR();
#endif
        __syncthreads();

        if (kb <= qbase + 15) {   // wave has at least one unmasked key here
            // S = (Q/sqrtD) K^T : two 16x16 subtiles, 2 WMMAs each
            v8f sa0 = {}, sa1 = {};
            {
                v16h bk0 = frag_b(Klds, 72);
                v16h bk1 = frag_b(Klds + 32, 72);
                v16h bk2 = frag_b(Klds + 16 * 72, 72);
                v16h bk3 = frag_b(Klds + 16 * 72 + 32, 72);
                sa0 = wmma16(qf0, bk0, sa0);
                sa0 = wmma16(qf1, bk1, sa0);
                sa1 = wmma16(qf0, bk2, sa1);
                sa1 = wmma16(qf1, bk3, sa1);
            }

            if (kb + 31 <= qbase) {
                // Interior block: fully unmasked, no guards
#pragma unroll
                for (int r = 0; r < 8; ++r) {
                    float mx = fmaxf(sa0[r], sa1[r]);
#pragma unroll
                    for (int msk = 1; msk < 16; msk <<= 1)
                        mx = fmaxf(mx, __shfl_xor(mx, msk, 32));
                    const float mnew  = fmaxf(mrow[r], mx);
                    const float alpha = __expf(mrow[r] - mnew);
                    const float p0 = __expf(sa0[r] - mnew);
                    const float p1 = __expf(sa1[r] - mnew);
                    mrow[r] = mnew;
                    lrow[r] *= alpha;
#pragma unroll
                    for (int dt = 0; dt < 4; ++dt) acc[dt][r] *= alpha;
                    const int prow = r + 8 * hs;
                    Plds[wave][prow * 40 + n]      = (half_t)p0;
                    Plds[wave][prow * 40 + 16 + n] = (half_t)p1;
                }
            } else {
                // Diagonal block: causal mask + masked-exp guards
#pragma unroll
                for (int r = 0; r < 8; ++r) {
                    const int qrow = qbase + r + 8 * hs;
                    const float s0 = (kb + n <= qrow)      ? sa0[r] : -3.0e38f;
                    const float s1 = (kb + 16 + n <= qrow) ? sa1[r] : -3.0e38f;
                    float mx = fmaxf(s0, s1);
#pragma unroll
                    for (int msk = 1; msk < 16; msk <<= 1)
                        mx = fmaxf(mx, __shfl_xor(mx, msk, 32));
                    const float mnew  = fmaxf(mrow[r], mx);
                    const float alpha = __expf(mrow[r] - mnew);
                    const float p0 = (s0 < -1e37f) ? 0.f : __expf(s0 - mnew);
                    const float p1 = (s1 < -1e37f) ? 0.f : __expf(s1 - mnew);
                    mrow[r] = mnew;
                    lrow[r] *= alpha;
#pragma unroll
                    for (int dt = 0; dt < 4; ++dt) acc[dt][r] *= alpha;
                    const int prow = r + 8 * hs;
                    Plds[wave][prow * 40 + n]      = (half_t)p0;
                    Plds[wave][prow * 40 + 16 + n] = (half_t)p1;
                }
            }
            asm volatile("s_wait_dscnt 0" ::: "memory");  // P visible to own wave

            v16h pf = frag_a(&Plds[wave][0], 40);

            // Row-sum of P in one WMMA: P(16x32) x ones(32x16)
            v8f zero = {};
            v8f lsum = wmma16(pf, onesf, zero);
#pragma unroll
            for (int r = 0; r < 8; ++r) lrow[r] += lsum[r];

            // O += P V
#pragma unroll
            for (int dt = 0; dt < 4; ++dt) {
                v16h bv = frag_b(VldsT + (dt * 16) * 40, 40);
                acc[dt] = wmma16(pf, bv, acc[dt]);
            }
        }
        __syncthreads();
    }

    // Epilogue: y[B,T,H*D] f16 feed for projection GEMM
    const size_t outbase = ((size_t)b * SEQ) * EMB + (size_t)h * HD;
#pragma unroll
    for (int dt = 0; dt < 4; ++dt)
#pragma unroll
        for (int r = 0; r < 8; ++r) {
            const int t = qbase + r + 8 * hs;
            yh[outbase + (size_t)t * EMB + dt * 16 + n] =
                (half_t)(acc[dt][r] / lrow[r]);
        }
}

// ------------------------------- launcher -----------------------------------
extern "C" void kernel_launch(void* const* d_in, const int* in_sizes, int n_in,
                              void* d_out, int out_size, void* d_ws, size_t ws_size,
                              hipStream_t stream) {
    const float* x      = (const float*)d_in[0];   // [4,2048,1024]
    const float* W_qkv  = (const float*)d_in[1];   // [3072,1024]
    const float* W_proj = (const float*)d_in[2];   // [1024,1024]
    float* out = (float*)d_out;

    const size_t M = (size_t)BATCH * SEQ;          // 8192
    char* ws = (char*)d_ws;
    half_t* xh    = (half_t*)ws;                 ws += M * EMB * 2;
    half_t* wqkvh = (half_t*)ws;                 ws += (size_t)3 * EMB * EMB * 2;
    half_t* wprjh = (half_t*)ws;                 ws += (size_t)EMB * EMB * 2;
    half_t* qkvh  = (half_t*)ws;                 ws += M * 3 * EMB * 2;
    half_t* qh    = (half_t*)ws;                 ws += M * EMB * 2;
    half_t* kh    = (half_t*)ws;                 ws += M * EMB * 2;
    half_t* vh    = (half_t*)ws;                 ws += M * EMB * 2;
    half_t* yh    = (half_t*)ws;                 ws += M * EMB * 2;

    // 1) fp32 -> f16 operands
    cvt_f32_f16<<<2048, 256, 0, stream>>>(x, xh, (int)(M * EMB));
    cvt_f32_f16<<<1024, 256, 0, stream>>>(W_qkv, wqkvh, 3 * EMB * EMB);
    cvt_f32_f16<<<512, 256, 0, stream>>>(W_proj, wprjh, EMB * EMB);

    // 2) qkv = x @ W_qkv^T   (M=8192, N=3072, K=1024)
    wmma_gemm_nt<true><<<(8192 / 128) * (3072 / 256), 256, 0, stream>>>(
        xh, wqkvh, nullptr, qkvh, 8192, 3072, 1024);

    // 3) RoPE + split to [B,H,T,D]
    rope_split<<<(BATCH * NH * SEQ * (HD / 2)) / 256, 256, 0, stream>>>(
        qkvh, qh, kh, vh);

    // 4) causal flash attention -> yh [B,T,EMB] f16
    flash_attn<<<BATCH * NH * (SEQ / 64), 128, 0, stream>>>(qh, kh, vh, yh);

    // 5) out = yh @ W_proj^T  (M=8192, N=1024, K=1024), f32 epilogue
    wmma_gemm_nt<false><<<(8192 / 128) * (1024 / 256), 256, 0, stream>>>(
        yh, wprjh, out, nullptr, 8192, 1024, 1024);
}
